// MiniGru_56315611185367
// MI455X (gfx1250) — matrix operand order repaired and checked by reading, not compiled, not run on hardware
//
#include <hip/hip_runtime.h>

// ---------------------------------------------------------------------------
// MiniGRU fused cell for gfx1250 (MI455X), bf16 WMMA path.
// ---------------------------------------------------------------------------

typedef __attribute__((ext_vector_type(16))) __bf16       v16bf;
typedef __attribute__((ext_vector_type(8)))  float        v8f;
typedef __attribute__((ext_vector_type(4)))  unsigned int v4u;

#define DETER   256
#define STOCH   128
#define ACT     16
#define HID     256
#define K1      160              // 144 zero-padded to 5 * 32
#define K2      512              // HID + DETER
#define N2      768              // 3 * DETER
#define BROWS   32               // rows per workgroup
#define A1_PITCH (K1 + 8)        // 336B row pitch -> bank-group rotation
#define A2_PITCH (K2 + 8)        // 1040B row pitch -> bank-group rotation

union FragU { v16bf bf; v4u u[2]; };

// Load one 16x32 bf16 fragment (16 bf16 per lane) as two 16B chunks.
__device__ __forceinline__ v16bf ld_frag(const __bf16* p, int off0, int off1) {
  FragU f;
  f.u[0] = *(const v4u*)(p + off0);
  f.u[1] = *(const v4u*)(p + off1);
  return f.bf;
}

__device__ __forceinline__ v8f wmma_bf16(v16bf a, v16bf b, v8f c) {
  return __builtin_amdgcn_wmma_f32_16x16x32_bf16(
      /*neg_a=*/false, a, /*neg_b=*/false, b,
      /*c_mod=*/(short)0, c, /*reuse_a=*/false, /*reuse_b=*/false);
}

// ---------------------------------------------------------------------------
// Pre-pass: fp32 weights -> bf16, transposed to N-major (K contiguous).
//   wCoreT: [N2][K2]   wProjT: [HID][K1] (K padded with zeros 144..159)
// ---------------------------------------------------------------------------
__global__ void cvt_weights(const float* __restrict__ proj_w,
                            const float* __restrict__ core_w,
                            __bf16* __restrict__ wProjT,
                            __bf16* __restrict__ wCoreT) {
  const int idx = blockIdx.x * 256 + threadIdx.x;
  const int CORE_N = N2 * K2;           // 393216
  const int PROJ_N = HID * K1;          // 40960
  if (idx < CORE_N) {
    const int n = idx / K2, k = idx - n * K2;
    wCoreT[idx] = (__bf16)core_w[k * N2 + n];
  } else {
    const int j = idx - CORE_N;
    if (j < PROJ_N) {
      const int n = j / K1, k = j - n * K1;
      wProjT[j] = (k < STOCH + ACT) ? (__bf16)proj_w[k * HID + n] : (__bf16)0.f;
    }
  }
}

// ---------------------------------------------------------------------------
// Fused: proj GEMM -> tokens -> core GEMM -> LayerNorm -> GRU gates.
// 256 threads = 8 wave32: wave = (mstrip<<2)|nchunk
//   mstrip (0,1): 16-row strip; nchunk (0..3): 64 base columns.
// Each wave owns gate columns {n, n+256, n+512} for its base n.
// ---------------------------------------------------------------------------
extern "C" __global__ void __launch_bounds__(256)
minigru_fused(const float*  __restrict__ deter,
              const float*  __restrict__ stoch,
              const float*  __restrict__ action,
              const float*  __restrict__ proj_b,
              const __bf16* __restrict__ wProjT,
              const __bf16* __restrict__ wCoreT,
              const float*  __restrict__ ln_g,
              const float*  __restrict__ ln_b,
              float*        __restrict__ out) {
  __shared__ __bf16 a1[BROWS * A1_PITCH];
  __shared__ __bf16 a2[BROWS * A2_PITCH];
  __shared__ float  red[2 * BROWS];     // [0,32): sum   [32,64): sumsq

  const int tid  = threadIdx.x;
  const int wave = tid >> 5;
  const int lane = tid & 31;
  const int nlo  = lane & 15;
  const int half = lane >> 4;
  const int m0   = (wave >> 2) << 4;    // 0 or 16
  const int nch  = wave & 3;            // 64-col chunk
  const int rowBase = blockIdx.x * BROWS;

  if (tid < 2 * BROWS) red[tid] = 0.f;

  // ---- stage A1 = [stoch | action | zeros] (bf16) ----
  for (int idx = tid; idx < BROWS * K1; idx += 256) {
    const int r = idx / K1, c = idx - r * K1;
    const int gr = rowBase + r;
    float v = 0.f;
    if (c < STOCH)            v = stoch[gr * STOCH + c];
    else if (c < STOCH + ACT) v = action[gr * ACT + (c - STOCH)];
    a1[r * A1_PITCH + c] = (__bf16)v;
  }
  // ---- stage deter into right half of A2 (K = 256..511) ----
  for (int idx = tid; idx < BROWS * DETER; idx += 256) {
    const int r = idx >> 8, c = idx & (DETER - 1);
    a2[r * A2_PITCH + HID + c] = (__bf16)deter[(rowBase + r) * DETER + c];
  }
  __syncthreads();

  // Fragment K-offsets per ISA VGPR layouts (wave32):
  //   A 16x32 16-bit: elems 0..7 -> K = 8*half + e ; 8..15 -> K = 16 + 8*half + e
  //   B 32x16 16-bit: lane = N, elems 0..7 -> K = 8*half + e ; 8..15 -> K = 16 + 8*half + e
  const int aoff0 = 8 * half,  aoff1 = 16 + 8 * half;
  const int boff0 = 8 * half, boff1 = 16 + 8 * half;

  // ---- GEMM1: tokens[32,256] ----
  v8f acc1[4] = {};
  for (int kt = 0; kt < K1 / 32; ++kt) {
    const v16bf af = ld_frag(&a1[(m0 + nlo) * A1_PITCH + kt * 32], aoff0, aoff1);
#pragma unroll
    for (int nt = 0; nt < 4; ++nt) {
      const int n = nch * 64 + nt * 16 + nlo;
      const v16bf bf = ld_frag(&wProjT[n * K1 + kt * 32], boff0, boff1);
      acc1[nt] = wmma_bf16(af, bf, acc1[nt]);
    }
  }
  // epilogue: + proj_b, cvt to bf16 into left half of A2 (K = 0..255)
#pragma unroll
  for (int nt = 0; nt < 4; ++nt) {
    const int n = nch * 64 + nt * 16 + nlo;
    const float pb = proj_b[n];
#pragma unroll
    for (int r = 0; r < 8; ++r) {
      const int row = m0 + r + 8 * half;      // C/D layout: M = r + 8*half
      a2[row * A2_PITCH + n] = (__bf16)(acc1[nt][r] + pb);
    }
  }
  __syncthreads();

  // ---- GEMM2: parts[32,768], gate-aligned column ownership ----
  v8f acc[3][4] = {};
  for (int kt = 0; kt < K2 / 32; ++kt) {
    const v16bf af = ld_frag(&a2[(m0 + nlo) * A2_PITCH + kt * 32], aoff0, aoff1);
#pragma unroll
    for (int g = 0; g < 3; ++g)
#pragma unroll
      for (int nt = 0; nt < 4; ++nt) {
        const int n = g * DETER + nch * 64 + nt * 16 + nlo;
        const v16bf bf = ld_frag(&wCoreT[n * K2 + kt * 32], boff0, boff1);
        acc[g][nt] = wmma_bf16(af, bf, acc[g][nt]);
      }
  }

  // ---- LayerNorm partials: per-lane -> 16-lane shuffle -> LDS atomics ----
  float s[8], ss[8];
#pragma unroll
  for (int r = 0; r < 8; ++r) { s[r] = 0.f; ss[r] = 0.f; }
#pragma unroll
  for (int g = 0; g < 3; ++g)
#pragma unroll
    for (int nt = 0; nt < 4; ++nt)
#pragma unroll
      for (int r = 0; r < 8; ++r) {
        const float v = acc[g][nt][r];
        s[r] += v; ss[r] += v * v;
      }
#pragma unroll
  for (int m = 1; m < 16; m <<= 1)
#pragma unroll
    for (int r = 0; r < 8; ++r) {
      s[r]  += __shfl_xor(s[r],  m, 32);
      ss[r] += __shfl_xor(ss[r], m, 32);
    }
  if (nlo == 0) {
#pragma unroll
    for (int r = 0; r < 8; ++r) {
      const int row = m0 + r + 8 * half;
      atomicAdd(&red[row], s[r]);
      atomicAdd(&red[BROWS + row], ss[r]);
    }
  }
  __syncthreads();

  // ---- LN + gates + store ----
#pragma unroll
  for (int nt = 0; nt < 4; ++nt) {
    const int ncol = nch * 64 + nt * 16 + nlo;
    const float g0 = ln_g[ncol],             b0 = ln_b[ncol];
    const float g1 = ln_g[DETER + ncol],     b1 = ln_b[DETER + ncol];
    const float g2 = ln_g[2 * DETER + ncol], b2 = ln_b[2 * DETER + ncol];
#pragma unroll
    for (int r = 0; r < 8; ++r) {
      const int row  = m0 + r + 8 * half;
      const float mean = red[row] * (1.f / 768.f);
      const float var  = red[BROWS + row] * (1.f / 768.f) - mean * mean;
      const float inv  = __frsqrt_rn(var + 1e-5f);
      const float xr = (acc[0][nt][r] - mean) * inv * g0 + b0;
      const float xc = (acc[1][nt][r] - mean) * inv * g1 + b1;
      const float xu = (acc[2][nt][r] - mean) * inv * g2 + b2;
      const float rst  = 1.f / (1.f + __expf(-xr));
      const float cand = tanhf(rst * xc);
      const float upd  = 1.f / (1.f + __expf(-(xu - 1.f)));
      const int gidx = (rowBase + row) * DETER + ncol;
      out[gidx] = upd * cand + (1.f - upd) * deter[gidx];
    }
  }
}

// ---------------------------------------------------------------------------
extern "C" void kernel_launch(void* const* d_in, const int* in_sizes, int n_in,
                              void* d_out, int out_size, void* d_ws, size_t ws_size,
                              hipStream_t stream) {
  const float* deter  = (const float*)d_in[0];
  const float* stoch  = (const float*)d_in[1];
  const float* action = (const float*)d_in[2];
  const float* proj_w = (const float*)d_in[3];
  const float* proj_b = (const float*)d_in[4];
  const float* core_w = (const float*)d_in[5];
  const float* ln_g   = (const float*)d_in[6];
  const float* ln_b   = (const float*)d_in[7];
  float* out = (float*)d_out;

  // workspace: core_wT bf16 [768][512] then proj_wT bf16 [256][160] (~850 KB)
  __bf16* wCoreT = (__bf16*)d_ws;
  __bf16* wProjT = wCoreT + (size_t)N2 * K2;

  const int cvt_elems = N2 * K2 + HID * K1;
  cvt_weights<<<(cvt_elems + 255) / 256, 256, 0, stream>>>(proj_w, core_w, wProjT, wCoreT);

  const int B = in_sizes[0] / DETER;    // 131072
  minigru_fused<<<B / BROWS, 256, 0, stream>>>(
      deter, stoch, action, proj_b, wProjT, wCoreT, ln_g, ln_b, out);
}
